// LlamaAttention_60748017435038
// MI455X (gfx1250) — matrix operand order repaired and checked by reading, not compiled, not run on hardware
//
#include <hip/hip_runtime.h>

typedef unsigned short u16;
typedef __bf16 bf16_t;
typedef bf16_t v16bf __attribute__((ext_vector_type(16)));
typedef float  v8f   __attribute__((ext_vector_type(8)));
typedef unsigned int v4u  __attribute__((ext_vector_type(4)));
typedef unsigned int v4ui __attribute__((ext_vector_type(4)));
typedef int          v8i  __attribute__((ext_vector_type(8)));
typedef int          v4i  __attribute__((ext_vector_type(4)));

struct Frag2 { v4u a, b; };

static __device__ __forceinline__ v16bf load_frag(const u16* p0, const u16* p1) {
  Frag2 f;
  f.a = *(const v4u*)p0;
  f.b = *(const v4u*)p1;
  return __builtin_bit_cast(v16bf, f);
}

static __device__ __forceinline__ u16 f2bf(float f) {
  unsigned u = __float_as_uint(f);
  unsigned r = u + 0x7FFFu + ((u >> 16) & 1u);
  return (u16)(r >> 16);
}
static __device__ __forceinline__ float bf2f(u16 h) {
  return __uint_as_float(((unsigned)h) << 16);
}

// ---------------------------------------------------------------------------
// TDM: DMA a 2-D bf16 tile (tile_dim0=64 cols, `rows` rows, row stride K)
// from global into LDS, padding 16B after every 128B row (LDS row stride =
// 144B = 72 halves).  D# layout per CDNA5 ISA ch.8.  This toolchain's
// builtin takes 6 args: (g0, g1, g2, g3, extra, cpol); groups 2/3 and the
// extra slot are zero for a 2-D tile.
// ---------------------------------------------------------------------------
static __device__ __forceinline__ void tdm_load_tile(
    const u16* gsrc, unsigned lds_addr, int rows, int K)
{
  unsigned long long ga = (unsigned long long)(uintptr_t)gsrc;
  v4ui g0;
  g0.x = 1u;                                        // count=1, user descriptor
  g0.y = lds_addr;                                  // lds_addr[31:0]
  g0.z = (unsigned)(ga & 0xFFFFFFFFull);            // global_addr[31:0]
  g0.w = (unsigned)((ga >> 32) & 0x1FFFFFFull)      // global_addr[56:32]
       | (2u << 30);                                // type = 2 ("image")
  v8i g1;
  g1[0] = (int)((1u << 16)      // data_size = 2 bytes
              | (1u << 20)      // pad_enable
              | (4u << 22)      // pad_interval: 32 DWORDs (128B) between pads
              | (3u << 25));    // pad_amount:   4 DWORDs (16B)
  g1[1] = (int)(((unsigned)K & 0xFFFFu) << 16);                   // tensor_dim0 lo
  g1[2] = (int)(((unsigned)K >> 16) | (((unsigned)rows & 0xFFFFu) << 16)); // td0 hi | td1 lo
  g1[3] = (int)(64u << 16);                                       // td1 hi=0 | tile_dim0=64
  g1[4] = (int)((unsigned)rows & 0xFFFFu);                        // tile_dim1=rows, tile_dim2=0
  g1[5] = (int)(unsigned)K;                                       // tensor_dim0_stride[31:0]
  g1[6] = 0;                                                      // stride hi | td1_stride lo
  g1[7] = 0;
  v4i z4 = {0, 0, 0, 0};
  v8i z8 = {0, 0, 0, 0, 0, 0, 0, 0};
  __builtin_amdgcn_tensor_load_to_lds(g0, g1, z4, z4, z8, 0);
}

// ---------------------------------------------------------------------------
// fp32 -> bf16 conversion
// ---------------------------------------------------------------------------
__global__ __launch_bounds__(256) void cvt_bf16_kernel(const float* __restrict__ src,
                                                       u16* __restrict__ dst, int n) {
  int i = blockIdx.x * 256 + threadIdx.x;
  if (i < n) dst[i] = f2bf(src[i]);
}

// ---------------------------------------------------------------------------
// C[M,N] = A[M,K] * W[N,K]^T   (bf16 in, f32 accum, bf16 or f32 out)
// Block: 256 threads (8 waves, 2x4), tile 64(M) x 128(N), BK = 64.
// Double-buffered LDS fed by the Tensor Data Mover (leader wave issues the
// DMA, s_wait_tensorcnt paces the pipeline).
// ---------------------------------------------------------------------------
template<bool OUT_BF16>
__global__ __launch_bounds__(256) void gemm_bf16_kernel(
    const u16* __restrict__ A, const u16* __restrict__ W,
    void* __restrict__ Cout, int M, int N, int K)
{
  __shared__ u16 lA[2][64 * 72];    // 64 rows * (64 + 8 pad) halves
  __shared__ u16 lB[2][128 * 72];

  const int tid  = threadIdx.x;
  const int wave = tid >> 5;
  const int lane = tid & 31;
  const int l15  = lane & 15;
  const int lh   = lane >> 4;
  const int wm   = wave >> 2;   // 0..1
  const int wn   = wave & 3;    // 0..3
  const int bm0  = blockIdx.y * 64;
  const int bn0  = blockIdx.x * 128;
  const bool leader = (wave == 0);

  const v8f vzero = {0.f,0.f,0.f,0.f,0.f,0.f,0.f,0.f};
  v8f acc[2][2];
  #pragma unroll
  for (int i = 0; i < 2; ++i)
    #pragma unroll
    for (int j = 0; j < 2; ++j)
      acc[i][j] = vzero;

  // Prologue: stage 0 into buffer 0.
  if (leader) {
    tdm_load_tile(&A[(size_t)bm0 * K], (unsigned)(uintptr_t)&lA[0][0], 64,  K);
    tdm_load_tile(&W[(size_t)bn0 * K], (unsigned)(uintptr_t)&lB[0][0], 128, K);
  }

  const int nks = K >> 6;
  for (int ks = 0; ks < nks; ++ks) {
    const int cur = ks & 1;
    if (leader) {
      if (ks + 1 < nks) {
        const int k1 = (ks + 1) << 6;
        tdm_load_tile(&A[(size_t)bm0 * K + k1], (unsigned)(uintptr_t)&lA[cur ^ 1][0], 64,  K);
        tdm_load_tile(&W[(size_t)bn0 * K + k1], (unsigned)(uintptr_t)&lB[cur ^ 1][0], 128, K);
        __builtin_amdgcn_s_wait_tensorcnt(2);   // in-order: stage `ks` landed
      } else {
        __builtin_amdgcn_s_wait_tensorcnt(0);
      }
    }
    __syncthreads();                             // tile `cur` visible to all waves

    v16bf fa[2][2], fbm[2][2];
    #pragma unroll
    for (int i = 0; i < 2; ++i)
      #pragma unroll
      for (int s2 = 0; s2 < 2; ++s2) {
        const u16* p = &lA[cur][(wm * 32 + i * 16 + l15) * 72 + s2 * 32 + lh * 8];
        fa[i][s2] = load_frag(p, p + 16);        // A: K chunks {lh*8, 16+lh*8}
      }
    #pragma unroll
    for (int j = 0; j < 2; ++j)
      #pragma unroll
      for (int s2 = 0; s2 < 2; ++s2) {
        const u16* p = &lB[cur][(wn * 32 + j * 16 + l15) * 72 + s2 * 32 + lh * 8];
        fbm[j][s2] = load_frag(p, p + 16);       // B: K chunks {lh*8, 16+lh*8}
      }
    #pragma unroll
    for (int i = 0; i < 2; ++i)
      #pragma unroll
      for (int j = 0; j < 2; ++j)
        #pragma unroll
        for (int s2 = 0; s2 < 2; ++s2)
          acc[i][j] = __builtin_amdgcn_wmma_f32_16x16x32_bf16(
              false, fa[i][s2], false, fbm[j][s2], (short)0, acc[i][j], false, false);
    __syncthreads();                             // done reading before overwrite
  }

  #pragma unroll
  for (int i = 0; i < 2; ++i) {
    #pragma unroll
    for (int j = 0; j < 2; ++j) {
      const int row0 = bm0 + wm * 32 + i * 16 + lh * 8;
      const int col  = bn0 + wn * 32 + j * 16 + l15;
      if (OUT_BF16) {
        u16* C = (u16*)Cout;
        #pragma unroll
        for (int r = 0; r < 8; ++r)
          C[(size_t)(row0 + r) * N + col] = f2bf(acc[i][j][r]);
      } else {
        float* C = (float*)Cout;
        #pragma unroll
        for (int r = 0; r < 8; ++r)
          C[(size_t)(row0 + r) * N + col] = acc[i][j][r];
      }
    }
  }
}

// ---------------------------------------------------------------------------
// RoPE in place on bf16 Q or K.  X: (4096 tokens, nheads*64); one thread per
// rotation pair (d, d+32).  cos/sin: (B*S, 64) fp32.
// ---------------------------------------------------------------------------
__global__ __launch_bounds__(256) void rope_kernel(
    u16* __restrict__ X, const float* __restrict__ Cos,
    const float* __restrict__ Sin, int nheads)
{
  const int idx  = blockIdx.x * 256 + threadIdx.x;
  const int d    = idx & 31;
  const int rest = idx >> 5;
  const int h    = rest % nheads;
  const int tok  = rest / nheads;
  const float c1 = Cos[tok * 64 + d];
  const float s1 = Sin[tok * 64 + d];
  const float c2 = Cos[tok * 64 + d + 32];
  const float s2 = Sin[tok * 64 + d + 32];
  const size_t base = (size_t)tok * (nheads * 64) + h * 64 + d;
  const float x1 = bf2f(X[base]);
  const float x2 = bf2f(X[base + 32]);
  X[base]      = f2bf(x1 * c1 - x2 * s1);
  X[base + 32] = f2bf(x2 * c2 + x1 * s2);
}

// ---------------------------------------------------------------------------
// V (token, kvh*64+d) -> Vt (b, kvh, d, s): key dimension becomes contiguous.
// ---------------------------------------------------------------------------
__global__ __launch_bounds__(256) void vtrans_kernel(const u16* __restrict__ V,
                                                     u16* __restrict__ Vt)
{
  const int idx = blockIdx.x * 256 + threadIdx.x;   // 2*2048*512 total
  const int c   = idx & 511;                        // kvh*64 + d
  const int tok = idx >> 9;
  const int b   = tok >> 11;
  const int s   = tok & 2047;
  Vt[(size_t)(b * 512 + c) * 2048 + s] = V[idx];
}

// ---------------------------------------------------------------------------
// Flash attention: one wave per (b, h, 16-query tile).  Causal, online
// softmax, GQA (4 heads share one kv head).  Scores + P.V via WMMA bf16.
// ---------------------------------------------------------------------------
__global__ __launch_bounds__(256) void attn_kernel(
    const u16* __restrict__ Q, const u16* __restrict__ Kc,
    const u16* __restrict__ Vt, u16* __restrict__ Out)
{
  __shared__ u16 ldsP[8][16 * 40];
  const int wave = threadIdx.x >> 5;
  const int lane = threadIdx.x & 31;
  const int l15  = lane & 15;
  const int lh   = lane >> 4;
  const int gid  = blockIdx.x * 8 + wave;   // 8192 (b,h,qt) tiles
  const int b    = gid >> 12;               // NH*128 = 4096 per batch
  const int rem  = gid & 4095;
  const int h    = rem >> 7;
  const int qt   = rem & 127;
  const int kvh  = h >> 2;                  // GROUPS = 4
  const int q0   = qt * 16;
  const int tok0 = b * 2048;

  // Q tile 16x64 as two A fragments (K = hd 0..31, 32..63), held in regs.
  v16bf aq[2];
  {
    const u16* qrowp = &Q[(size_t)(tok0 + q0 + l15) * 2048 + h * 64];
    #pragma unroll
    for (int s2 = 0; s2 < 2; ++s2) {
      const u16* p = qrowp + s2 * 32 + lh * 8;
      aq[s2] = load_frag(p, p + 16);
    }
  }

  const v8f vzero = {0.f,0.f,0.f,0.f,0.f,0.f,0.f,0.f};
  v8f o[4] = {vzero, vzero, vzero, vzero};
  float mrow[8], lrow[8];
  #pragma unroll
  for (int r = 0; r < 8; ++r) { mrow[r] = -3.0e38f; lrow[r] = 0.f; }

  const int nk = q0 + 16;             // causal: keys 0..q0+15 visible
  u16* myP = ldsP[wave];

  for (int kb = 0; kb < nk; kb += 32) {
    // ---- scores: two 16x16 tiles over keys kb..kb+31 -----------------
    v8f sc0 = vzero, sc1 = vzero;
    #pragma unroll
    for (int s2 = 0; s2 < 2; ++s2) {
      const u16* p = &Kc[(size_t)(tok0 + kb + l15) * 512 + kvh * 64 + s2 * 32 + lh * 8];
      v16bf fk = load_frag(p, p + 16);
      sc0 = __builtin_amdgcn_wmma_f32_16x16x32_bf16(
          false, aq[s2], false, fk, (short)0, sc0, false, false);
    }
    const bool t1 = (kb + 16) < nk;
    if (t1) {
      #pragma unroll
      for (int s2 = 0; s2 < 2; ++s2) {
        const u16* p = &Kc[(size_t)(tok0 + kb + 16 + l15) * 512 + kvh * 64 + s2 * 32 + lh * 8];
        v16bf fk = load_frag(p, p + 16);
        sc1 = __builtin_amdgcn_wmma_f32_16x16x32_bf16(
            false, aq[s2], false, fk, (short)0, sc1, false, false);
      }
    }

    // ---- causal mask + online softmax over this 32-key block ---------
    float alpha[8];
    #pragma unroll
    for (int r = 0; r < 8; ++r) {
      const int qrow = q0 + lh * 8 + r;
      float v0 = (kb + l15 <= qrow) ? sc0[r] * 0.125f : -1e30f;
      float v1 = (t1 && (kb + 16 + l15 <= qrow)) ? sc1[r] * 0.125f : -1e30f;
      float mx = fmaxf(v0, v1);
      #pragma unroll
      for (int msk = 1; msk < 16; msk <<= 1)
        mx = fmaxf(mx, __shfl_xor(mx, msk, 32));
      const float mnew = fmaxf(mrow[r], mx);
      const float p0 = __expf(v0 - mnew);
      const float p1 = __expf(v1 - mnew);
      float ps = p0 + p1;
      #pragma unroll
      for (int msk = 1; msk < 16; msk <<= 1)
        ps += __shfl_xor(ps, msk, 32);
      alpha[r] = __expf(mrow[r] - mnew);
      lrow[r]  = lrow[r] * alpha[r] + ps;
      mrow[r]  = mnew;
      // P (C-layout) -> LDS, to be re-read in A-operand layout.
      myP[(lh * 8 + r) * 40 + l15]      = f2bf(p0);
      myP[(lh * 8 + r) * 40 + 16 + l15] = f2bf(p1);
    }
    asm volatile("s_wait_dscnt 0x0" ::: "memory");

    // ---- rescale O, then O += P . V ----------------------------------
    #pragma unroll
    for (int nt = 0; nt < 4; ++nt)
      #pragma unroll
      for (int r = 0; r < 8; ++r)
        o[nt][r] *= alpha[r];

    const u16* pp = &myP[l15 * 40 + lh * 8];
    v16bf ap = load_frag(pp, pp + 16);   // P as 16x32 A fragment
    #pragma unroll
    for (int nt = 0; nt < 4; ++nt) {
      const u16* vp = &Vt[(size_t)((b * 8 + kvh) * 64 + nt * 16 + l15) * 2048 + kb + lh * 8];
      v16bf fv = load_frag(vp, vp + 16); // keys contiguous thanks to Vt
      o[nt] = __builtin_amdgcn_wmma_f32_16x16x32_bf16(
          false, ap, false, fv, (short)0, o[nt], false, false);
    }
  }

  float inv[8];
  #pragma unroll
  for (int r = 0; r < 8; ++r) inv[r] = 1.f / lrow[r];
  #pragma unroll
  for (int nt = 0; nt < 4; ++nt) {
    #pragma unroll
    for (int r = 0; r < 8; ++r) {
      const int row = tok0 + q0 + lh * 8 + r;
      const int col = h * 64 + nt * 16 + l15;
      Out[(size_t)row * 2048 + col] = f2bf(o[nt][r] * inv[r]);
    }
  }
}

// ---------------------------------------------------------------------------
extern "C" void kernel_launch(void* const* d_in, const int* in_sizes, int n_in,
                              void* d_out, int out_size, void* d_ws, size_t ws_size,
                              hipStream_t stream)
{
  const float* hs = (const float*)d_in[0];
  const float* cs = (const float*)d_in[1];
  const float* sn = (const float*)d_in[2];
  const float* wq = (const float*)d_in[3];
  const float* wk = (const float*)d_in[4];
  const float* wv = (const float*)d_in[5];
  const float* wo = (const float*)d_in[6];

  // Workspace layout (bf16 elements): ~80 MiB total.
  u16* w    = (u16*)d_ws;
  u16* hsb  = w;  w += 8388608;   // hidden  4096x2048
  u16* wqb  = w;  w += 4194304;   // wq      2048x2048
  u16* wkb  = w;  w += 1048576;   // wk       512x2048
  u16* wvb  = w;  w += 1048576;   // wv       512x2048
  u16* wob  = w;  w += 4194304;   // wo      2048x2048
  u16* qb   = w;  w += 8388608;   // q       4096x2048
  u16* kbuf = w;  w += 2097152;   // k       4096x512
  u16* vbuf = w;  w += 2097152;   // v       4096x512
  u16* vtb  = w;  w += 2097152;   // v^T     (B,8,64,2048)
  u16* aob  = w;  w += 8388608;   // attn out 4096x2048

  cvt_bf16_kernel<<<dim3((8388608 + 255) / 256), dim3(256), 0, stream>>>(hs, hsb, 8388608);
  cvt_bf16_kernel<<<dim3((4194304 + 255) / 256), dim3(256), 0, stream>>>(wq, wqb, 4194304);
  cvt_bf16_kernel<<<dim3((1048576 + 255) / 256), dim3(256), 0, stream>>>(wk, wkb, 1048576);
  cvt_bf16_kernel<<<dim3((1048576 + 255) / 256), dim3(256), 0, stream>>>(wv, wvb, 1048576);
  cvt_bf16_kernel<<<dim3((4194304 + 255) / 256), dim3(256), 0, stream>>>(wo, wob, 4194304);

  // Q/K/V projections (TDM-fed, double-buffered WMMA GEMM).
  gemm_bf16_kernel<true><<<dim3(2048 / 128, 4096 / 64), dim3(256), 0, stream>>>(hsb, wqb, qb,   4096, 2048, 2048);
  gemm_bf16_kernel<true><<<dim3( 512 / 128, 4096 / 64), dim3(256), 0, stream>>>(hsb, wkb, kbuf, 4096,  512, 2048);
  gemm_bf16_kernel<true><<<dim3( 512 / 128, 4096 / 64), dim3(256), 0, stream>>>(hsb, wvb, vbuf, 4096,  512, 2048);

  // RoPE on Q (32 heads) and K (8 heads).
  rope_kernel<<<dim3(4194304 / 256), dim3(256), 0, stream>>>(qb,   cs, sn, 32);
  rope_kernel<<<dim3(1048576 / 256), dim3(256), 0, stream>>>(kbuf, cs, sn, 8);

  // V -> V^T for contiguous key-dimension B-operand loads.
  vtrans_kernel<<<dim3(2097152 / 256), dim3(256), 0, stream>>>(vbuf, vtb);

  // Flash attention: 8192 (b,h,qtile) waves, 8 per block.
  attn_kernel<<<dim3(1024), dim3(256), 0, stream>>>(qb, kbuf, vtb, aob);

  // Output projection, fp32 out.
  gemm_bf16_kernel<false><<<dim3(2048 / 128, 4096 / 64), dim3(256), 0, stream>>>(aob, wob, d_out, 4096, 2048, 2048);

  (void)in_sizes; (void)n_in; (void)out_size; (void)ws_size;
}
